// SeparateWindowedValuesCoordinatesAttention_22823456211092
// MI455X (gfx1250) — matrix operand order repaired and checked by reading, not compiled, not run on hardware
//
#include <hip/hip_runtime.h>
#include <math.h>

// ---------------- problem constants ----------------
#define B_     4
#define HT_    256
#define WS_    8
#define NH_    8
#define HD_    16
#define VD_    128
#define WH_    32
#define NTOK_  65536          // HT_*HT_

// ---------------- LDS layout (u16 strides, padded vs 64-bank conflicts) ----
#define XV_STR 132            // x staging: 64 x 128 bf16, padded
#define QK_STR 34             // per-head q/k: 64 x 32 bf16, padded
#define V_STR  18             // per-head v:   64 x 16 bf16, padded
#define P_STR  66             // per-wave softmax P: 64 x 64 bf16, padded
#define SMEM_BYTES 156608
// u16 offsets of the per-head arrays inside smem (contiguous: qcat,kcat,varr)
#define QKV_BASE_U16 33792    // byte 67584
#define QKV_STEP_U16 17408    // 34816 bytes = one 8-head q/k array

// ---------------- workspace layout (bytes) --------------------------------
#define YWS_BYTES   (B_ * NTOK_ * VD_ * 2)
#define WVP_U32     24576
#define WCP_U32     16384
#define WOP_U32     8192

typedef __attribute__((ext_vector_type(16))) __bf16 bf16x16;
typedef __attribute__((ext_vector_type(8)))  float  f32x8;

union FragU { unsigned u[8]; bf16x16 v; };

__device__ inline unsigned short bf16_1(float a) {
  union { __bf16 h; unsigned short s; } r; r.h = (__bf16)a; return r.s;
}
#if __has_builtin(__builtin_amdgcn_cvt_pk_bf16_f32)
typedef __attribute__((ext_vector_type(2))) __bf16 bf16x2;
__device__ inline unsigned pack_bf16_2(float a, float b) {
  union { bf16x2 h; unsigned u; } r;
  r.h = __builtin_amdgcn_cvt_pk_bf16_f32(a, b);
  return r.u;
}
#else
__device__ inline unsigned pack_bf16_2(float a, float b) {
  union { __bf16 h[2]; unsigned u; } r;
  r.h[0] = (__bf16)a; r.h[1] = (__bf16)b; return r.u;
}
#endif
__device__ inline f32x8 wmma_bf16(const FragU& a, const FragU& b, f32x8 c) {
  return __builtin_amdgcn_wmma_f32_16x16x32_bf16(false, a.v, false, b.v, (short)0, c, false, false);
}

// A-operand (16x32 bf16) fragment from an LDS tile stored row-major [row][k].
__device__ inline FragU lds_a_frag(const unsigned short* base, int stride,
                                   int m0, int k0, int lane) {
  FragU f;
  int hl = (lane >> 4) & 1;
  const unsigned short* rp = base + (m0 + (lane & 15)) * stride;
#pragma unroll
  for (int vg = 0; vg < 8; ++vg) {
    int k = k0 + hl * 16 + 2 * vg;
    f.u[vg] = *(const unsigned*)(rp + k);
  }
  return f;
}

// B-operand fragment from prepacked weights (two b128 loads).
__device__ inline FragU wpk_b_frag(const unsigned* wcol, int k0, int hl) {
  FragU f;
  const uint4* p = (const uint4*)(wcol + (k0 >> 1) + hl * 8);
  uint4 q0 = p[0], q1 = p[1];
  f.u[0] = q0.x; f.u[1] = q0.y; f.u[2] = q0.z; f.u[3] = q0.w;
  f.u[4] = q1.x; f.u[5] = q1.y; f.u[6] = q1.z; f.u[7] = q1.w;
  return f;
}

// ============ one-shot weight prepack: fp32 -> bf16 pairs, B-layout =======
__global__ __launch_bounds__(256) void prepack_kernel(
    const float* __restrict__ Wv, const float* __restrict__ Wc,
    const float* __restrict__ Wo, unsigned* __restrict__ wvp,
    unsigned* __restrict__ wcp, unsigned* __restrict__ wop) {
  int idx = blockIdx.x * 256 + threadIdx.x;   // one packed dword each
  if (idx < WVP_U32) {
    int colp = idx >> 6, kp = idx & 63;
    int kk = colp >> 7, t = colp & 127;
    int n = t * 3 + kk;
    wvp[idx] = pack_bf16_2(Wv[(2 * kp) * 384 + n], Wv[(2 * kp + 1) * 384 + n]);
  } else if (idx < WVP_U32 + WCP_U32) {
    int i2 = idx - WVP_U32;
    int colp = i2 >> 6, kp = i2 & 63;
    int kk = colp >> 7, t = colp & 127;
    int n = t * 2 + kk;
    wcp[i2] = pack_bf16_2(Wc[(2 * kp) * 256 + n], Wc[(2 * kp + 1) * 256 + n]);
  } else if (idx < WVP_U32 + WCP_U32 + WOP_U32) {
    int i2 = idx - WVP_U32 - WCP_U32;
    int n = i2 >> 6, kp = i2 & 63;
    wop[i2] = pack_bf16_2(Wo[(2 * kp) * 128 + n], Wo[(2 * kp + 1) * 128 + n]);
  }
}

// ================= fused projection + windowed attention =================
__global__ __launch_bounds__(256) void swin_attn_kernel(
    const float* __restrict__ ev, const float* __restrict__ ec,
    const unsigned* __restrict__ wvp, const float* __restrict__ bv,
    const unsigned* __restrict__ wcp, const float* __restrict__ bc,
    const float* __restrict__ pe, unsigned short* __restrict__ yws) {
  extern __shared__ char smem[];
  unsigned short* su16 = (unsigned short*)smem;
  unsigned short* xv   = su16;                                // 64 x XV_STR
  unsigned short* xc   = xv + 64 * XV_STR;                    // 64 x XV_STR
  unsigned short* Pb   = su16;                                // 8 x 64 x P_STR
  unsigned short* qcat = su16 + QKV_BASE_U16;                 // NH x 64 x QK_STR
  unsigned short* kcat = qcat + NH_ * 64 * QK_STR;            // NH x 64 x QK_STR
  unsigned short* varr = kcat + NH_ * 64 * QK_STR;            // NH x 64 x V_STR
  float*          pos  = (float*)(varr + NH_ * 64 * V_STR);   // 15 x 16 f32

  const int tid  = threadIdx.x;
  const int blk  = blockIdx.x;
  const int b    = blk >> 10;
  const int rem  = blk & 1023;
  const int wx   = rem >> 5, wy = rem & 31;
  const int lane = tid & 31, wave = tid >> 5, hl = lane >> 4;

  // ---- phase 0: stage rolled window tokens (fp32 -> bf16) into LDS ----
#pragma unroll
  for (int it = 0; it < 8; ++it) {
    int v  = tid + it * 256;
    int t  = v >> 5;
    int c4 = (v & 31) * 4;
    int w1 = t >> 3, w2 = t & 7;
    int gr = (wx * 8 + w1 + WS_ / 2) & 255;   // shifted-window roll
    int gc = (wy * 8 + w2 + WS_ / 2) & 255;
    size_t gt = (size_t)b * NTOK_ + (size_t)gr * 256 + gc;
    float4 fv = *(const float4*)(ev + gt * VD_ + c4);
    float4 fc = *(const float4*)(ec + gt * VD_ + c4);
    unsigned* pxv = (unsigned*)(xv + t * XV_STR + c4);
    pxv[0] = pack_bf16_2(fv.x, fv.y);  pxv[1] = pack_bf16_2(fv.z, fv.w);
    unsigned* pxc = (unsigned*)(xc + t * XV_STR + c4);
    pxc[0] = pack_bf16_2(fc.x, fc.y);  pxc[1] = pack_bf16_2(fc.z, fc.w);
  }
  if (tid < 225) pos[(tid / 15) * 16 + (tid % 15)] = pe[tid];
  __syncthreads();

  // ---- phase 1: projections, 160 16x16 tiles (96 v + 64 c) over K=128 ----
  for (int tile = wave; tile < 160; tile += 8) {
    bool isv = tile < 96;
    int m0, t0, kk, kmul;
    const unsigned* Wp;
    const float* bias;
    const unsigned short* xa;
    if (isv) { int s = tile % 24;  m0 = (tile / 24) * 16;
               kk = s >> 3; t0 = (s & 7) * 16;
               Wp = wvp; bias = bv; kmul = 3; xa = xv; }
    else     { int t2 = tile - 96; int s = t2 % 16; m0 = (t2 / 16) * 16;
               kk = s >> 3; t0 = (s & 7) * 16;
               Wp = wcp; bias = bc; kmul = 2; xa = xc; }
    const int t = t0 + (lane & 15);                  // e*NH + h
    const unsigned* wcol = Wp + (kk * 128 + t) * 64; // packed column base
    f32x8 acc = {};
#pragma unroll
    for (int k0 = 0; k0 < 128; k0 += 32) {
      FragU a  = lds_a_frag(xa, XV_STR, m0, k0, lane);
      FragU bf = wpk_b_frag(wcol, k0, hl);
      acc = wmma_bf16(a, bf, acc);
    }
    float bb = bias[t * kmul + kk];
    // fold dots scale 1/sqrt(HD)=0.25 into q at store time
    float scale = (kk == 0) ? 0.25f : 1.0f;
    // branchless scatter target: qcat/kcat/varr are contiguous in LDS
    const int hh = t & 7, e = t >> 3;
    const int dstr = (isv && kk == 2) ? V_STR : QK_STR;
    unsigned short* dstp = su16 + QKV_BASE_U16 + kk * QKV_STEP_U16
                         + (hh * 64 + m0) * dstr + (isv ? 0 : 16) + e;
#pragma unroll
    for (int r = 0; r < 8; ++r)
      dstp[(r + hl * 8) * dstr] = bf16_1((acc[r] + bb) * scale);
  }
  __syncthreads();   // also protects region0 reuse (x -> P)

  // ---- phase 2: attention for head h = wave ----
  const int h  = wave;
  const int jc = lane & 15;
  const unsigned short* qh = qcat + h * 64 * QK_STR;
  const unsigned short* kh = kcat + h * 64 * QK_STR;
  const unsigned short* vh = varr + h * 64 * V_STR;

  FragU aq[4];
#pragma unroll
  for (int ti = 0; ti < 4; ++ti) aq[ti] = lds_a_frag(qh, QK_STR, ti * 16, 0, lane);

  f32x8 d[4][4];
#pragma unroll
  for (int tj = 0; tj < 4; ++tj) {
    FragU bk;
    const unsigned short* kp = kh + (tj * 16 + jc) * QK_STR + hl * 16;
#pragma unroll
    for (int vg = 0; vg < 8; ++vg) bk.u[vg] = *(const unsigned*)(kp + 2 * vg);
#pragma unroll
    for (int ti = 0; ti < 4; ++ti) {
      f32x8 z = {};
      d[ti][tj] = wmma_bf16(aq[ti], bk, z);
    }
  }

  // relative position bias + shift masks (dots scale already folded into q)
  const bool lastx = (wx == WH_ - 1), lasty = (wy == WH_ - 1);
#pragma unroll
  for (int ti = 0; ti < 4; ++ti)
#pragma unroll
    for (int tj = 0; tj < 4; ++tj) {
      int jj = tj * 16 + jc;
      int xj = jj >> 3, yj = jj & 7;
#pragma unroll
      for (int r = 0; r < 8; ++r) {
        int ii = ti * 16 + r + hl * 8;
        int xi = ii >> 3, yi = ii & 7;
        float vv = d[ti][tj][r] + pos[(xj - xi + 7) * 16 + (yj - yi + 7)];
        if (lastx && ((ii >= 32) != (jj >= 32)))             vv -= 1e30f;
        if (lasty && (((ii & 7) >= 4) != ((jj & 7) >= 4)))   vv -= 1e30f;
        d[ti][tj][r] = vv;
      }
    }

  // softmax over j
#pragma unroll
  for (int ti = 0; ti < 4; ++ti) {
#pragma unroll
    for (int r = 0; r < 8; ++r) {
      float m = fmaxf(fmaxf(d[ti][0][r], d[ti][1][r]), fmaxf(d[ti][2][r], d[ti][3][r]));
#pragma unroll
      for (int msk = 1; msk < 16; msk <<= 1) m = fmaxf(m, __shfl_xor(m, msk, 32));
      float s = 0.f;
#pragma unroll
      for (int tj = 0; tj < 4; ++tj) {
        float e = __expf(d[ti][tj][r] - m);
        d[ti][tj][r] = e; s += e;
      }
#pragma unroll
      for (int msk = 1; msk < 16; msk <<= 1) s += __shfl_xor(s, msk, 32);
      float inv = 1.f / s;
#pragma unroll
      for (int tj = 0; tj < 4; ++tj) d[ti][tj][r] *= inv;
    }
  }

  // P (D layout) -> LDS -> reload as A operand for P@V
  unsigned short* Pw = Pb + wave * 64 * P_STR;
#pragma unroll
  for (int ti = 0; ti < 4; ++ti)
#pragma unroll
    for (int tj = 0; tj < 4; ++tj)
#pragma unroll
      for (int r = 0; r < 8; ++r)
        Pw[(ti * 16 + r + hl * 8) * P_STR + tj * 16 + jc] = bf16_1(d[ti][tj][r]);

  // V fragments (B operand, K = key token, N = head_dim e)
  FragU bvf[2];
#pragma unroll
  for (int kb = 0; kb < 2; ++kb) {
#pragma unroll
    for (int vg = 0; vg < 8; ++vg) {
      int j0 = kb * 32 + hl * 16 + 2 * vg;
      unsigned lo  = vh[j0 * V_STR + jc];
      unsigned hi2 = vh[(j0 + 1) * V_STR + jc];
      bvf[kb].u[vg] = lo | (hi2 << 16);
    }
  }

#pragma unroll
  for (int ti = 0; ti < 4; ++ti) {
    f32x8 acc = {};
#pragma unroll
    for (int kb = 0; kb < 2; ++kb) {
      FragU ap = lds_a_frag(Pw, P_STR, ti * 16, kb * 32, lane);
      acc = wmma_bf16(ap, bvf[kb], acc);
    }
#pragma unroll
    for (int r = 0; r < 8; ++r) {
      int ii = ti * 16 + r + hl * 8;
      int gr = wx * 8 + (ii >> 3), gc = wy * 8 + (ii & 7);  // no inverse roll
      size_t gt = (size_t)b * NTOK_ + (size_t)gr * 256 + gc;
      yws[gt * VD_ + h * HD_ + jc] = bf16_1(acc[r]);
    }
  }
}

// ================= output projection: out = y @ Wo + bo (fp32 out) ========
__global__ __launch_bounds__(256) void out_proj_kernel(
    const unsigned short* __restrict__ yws, const unsigned* __restrict__ wop,
    const float* __restrict__ bo, float* __restrict__ out) {
  const int lane = threadIdx.x & 31, wave = threadIdx.x >> 5, hl = lane >> 4;
  const int m0 = blockIdx.x * 16;
  const int n0 = wave * 16;
  const int mrow = m0 + (lane & 15);
  const int ncol = n0 + (lane & 15);
  const unsigned* wcol = wop + ncol * 64;
  f32x8 acc = {};
#pragma unroll
  for (int k0 = 0; k0 < 128; k0 += 32) {
    FragU a;
    int kb = k0 + hl * 16;
    uint4 q0 = *(const uint4*)(yws + (size_t)mrow * VD_ + kb);
    uint4 q1 = *(const uint4*)(yws + (size_t)mrow * VD_ + kb + 8);
    a.u[0] = q0.x; a.u[1] = q0.y; a.u[2] = q0.z; a.u[3] = q0.w;
    a.u[4] = q1.x; a.u[5] = q1.y; a.u[6] = q1.z; a.u[7] = q1.w;
    FragU bf = wpk_b_frag(wcol, k0, hl);
    acc = wmma_bf16(a, bf, acc);
  }
  float bb = bo[ncol];
#pragma unroll
  for (int r = 0; r < 8; ++r)
    out[(size_t)(m0 + r + hl * 8) * VD_ + ncol] = acc[r] + bb;
}

extern "C" void kernel_launch(void* const* d_in, const int* in_sizes, int n_in,
                              void* d_out, int out_size, void* d_ws, size_t ws_size,
                              hipStream_t stream) {
  const float* ev = (const float*)d_in[0];
  const float* ec = (const float*)d_in[1];
  const float* Wv = (const float*)d_in[2];
  const float* bv = (const float*)d_in[3];
  const float* Wc = (const float*)d_in[4];
  const float* bc = (const float*)d_in[5];
  const float* pe = (const float*)d_in[6];
  const float* Wo = (const float*)d_in[7];
  const float* bo = (const float*)d_in[8];

  unsigned short* yws = (unsigned short*)d_ws;               // 64 MB bf16
  unsigned* wvp = (unsigned*)((char*)d_ws + YWS_BYTES);
  unsigned* wcp = wvp + WVP_U32;
  unsigned* wop = wcp + WCP_U32;

  prepack_kernel<<<dim3((WVP_U32 + WCP_U32 + WOP_U32 + 255) / 256), dim3(256),
                   0, stream>>>(Wv, Wc, Wo, wvp, wcp, wop);

  (void)hipFuncSetAttribute((const void*)swin_attn_kernel,
                            hipFuncAttributeMaxDynamicSharedMemorySize, SMEM_BYTES);
  swin_attn_kernel<<<dim3(B_ * WH_ * WH_), dim3(256), SMEM_BYTES, stream>>>(
      ev, ec, wvp, bv, wcp, bc, pe, yws);
  out_proj_kernel<<<dim3((B_ * NTOK_) / 16), dim3(256), 0, stream>>>(
      yws, wop, bo, (float*)d_out);
}
